// FusedSequenceParallelAttn_58961311039917
// MI455X (gfx1250) — matrix operand order repaired and checked by reading, not compiled, not run on hardware
//
#include <hip/hip_runtime.h>

typedef _Float16 half_t;
typedef __attribute__((ext_vector_type(16))) _Float16 v16h;
typedef __attribute__((ext_vector_type(8)))  float    v8f;

#define HQ 32
#define HK 8
#define GQ 4
#define DH 128
#define WORLDSZ 8
#define RANKID 2
// 1/sqrt(128) * log2(e): base-2 softmax domain (saves a v_mul per v_exp)
#define SCALE_LOG2E (0.08838834764831845f * 1.4426950408889634f)
#define NEG_INF_F (-1e30f)
#define VT_STRIDE 40    // halves per padded row of transposed V (80B: 16B-aligned, banks spread)
#define SK_STRIDE 136   // halves per padded row of staged K   (272B: 16B-aligned, banks rotate by 4)

#if defined(__has_builtin)
#if __has_builtin(__builtin_amdgcn_exp2f)
#define EXP2F(x) __builtin_amdgcn_exp2f(x)
#endif
#if __has_builtin(__builtin_amdgcn_rcpf)
#define RCPF(x) __builtin_amdgcn_rcpf(x)
#endif
#if __has_builtin(__builtin_amdgcn_global_load_async_to_lds_b128) && \
    __has_builtin(__builtin_amdgcn_s_wait_asynccnt)
#define HAVE_ASYNC_LDS 1
#endif
#endif
#ifndef EXP2F
#define EXP2F(x) exp2f(x)
#endif
#ifndef RCPF
#define RCPF(x) (1.0f / (x))
#endif

// Types matching the builtin signature "vV4i*1V4i*3IiIi" revealed by the
// round-3 diagnostic: GCC-vector int4 pointers in AS1 (global) / AS3 (LDS).
typedef int gv4i __attribute__((vector_size(4 * sizeof(int))));
typedef __attribute__((address_space(1))) gv4i g1_v4i;
typedef __attribute__((address_space(3))) gv4i l3_v4i;

union Frag16 { uint4 u[2]; v16h h; };
union FragP  { unsigned u[8]; v16h h; };
union VLoad  { uint4 u[4]; half_t e[32]; };
union PackO  { uint4 u; half_t e[8]; };

static __device__ inline unsigned pack_h2(float a, float b) {
  union { _Float16 h[2]; unsigned u; } x;
  x.h[0] = (_Float16)a;
  x.h[1] = (_Float16)b;
  return x.u;
}

__global__ __launch_bounds__(128)
void zigzag_flash_attn_kernel(const half_t* __restrict__ q,
                              const half_t* __restrict__ k,
                              const half_t* __restrict__ v,
                              const int* __restrict__ cu_q,
                              const int* __restrict__ cu_k,
                              int nseq,
                              half_t* __restrict__ out) {
  __shared__ half_t sVt[DH * VT_STRIDE];  // V^T tile: [d][n], n = 0..31
  __shared__ half_t sK[32 * SK_STRIDE];   // K tile staged once per block: [n][d]

  const int tid  = threadIdx.x;
  const int lane = tid & 31;
  const int wave = tid >> 5;        // 0..3 : query head within GQA group
  const int m    = lane & 15;       // softmax row handled by this lane
  const int h    = lane >> 4;       // half-wave selector

  const int qtile   = blockIdx.x;
  const int kvh     = blockIdx.y;
  const int headq   = kvh * GQ + wave;
  const int q_start = qtile * 16;

  // sequence lookup
  int s = 0;
  while (s + 1 < nseq && q_start >= cu_q[s + 1]) ++s;
  const int q0 = cu_q[s];
  const int k0 = cu_k[s];
  const int Lk = cu_k[s + 1] - k0;
  const int c  = Lk / (2 * WORLDSZ);

  // zigzag causal positions: tile rows are one contiguous range (c % 16 == 0)
  const int ql0   = q_start - q0;
  const int pos0  = (ql0 < c) ? (ql0 + RANKID * c)
                              : (ql0 + (2 * WORLDSZ - 2 - RANKID) * c);
  const int pos_m = pos0 + m;                 // this lane's row frontier
  const int ktiles = (pos0 + 15 + 32) >> 5;   // ceil((pos0+16)/32)
  const int nfull  = (pos0 + 1) >> 5;         // tiles fully inside the causal region

  // ---- Q fragments: B operand of S^T WMMA (lane holds 16 contiguous dims)
  const half_t* qrow = q + ((size_t)(q_start + m) * HQ + headq) * DH;
  v16h qfrag[4];
#pragma unroll
  for (int dc = 0; dc < 4; ++dc) {
    Frag16 f;
    const uint4* p = (const uint4*)(qrow + dc * 32 + h * 16);
    f.u[0] = p[0];
    f.u[1] = p[1];
    qfrag[dc] = f.h;
  }

  v8f o[8];
#pragma unroll
  for (int t = 0; t < 8; ++t) o[t] = (v8f){};
  float row_m = NEG_INF_F;
  float row_l = 0.0f;

  const half_t* kbase = k + ((size_t)k0 * HK + kvh) * DH;
  const half_t* vbase = v + ((size_t)k0 * HK + kvh) * DH;

  for (int kt = 0; kt < ktiles; ++kt) {
    const int kb = kt * 32;

    // ---- prefetch next tile's K/V rows (global_prefetch_b8)
    if (kt + 1 < ktiles) {
      const int kb2 = kb + 32;
      __builtin_prefetch(kbase + (size_t)(kb2 + m) * (HK * DH), 0, 1);
      __builtin_prefetch(kbase + (size_t)(kb2 + 16 + m) * (HK * DH), 0, 1);
      __builtin_prefetch(vbase + (size_t)(kb2 + (tid & 31)) * (HK * DH), 0, 1);
    }

    __syncthreads();   // previous iteration's sVt/sK reads are done

    // ---- stage K tile (32 x 128) into LDS: async global->LDS (4x dedup across waves)
    {
#pragma unroll
      for (int i = 0; i < 4; ++i) {
        const int chunk = tid + 128 * i;        // 512 x 16B chunks
        const int row   = chunk >> 4;           // k position within tile
        const int off   = (chunk & 15) * 8;     // halves within the 128-dim row
        const half_t* gp = kbase + (size_t)(kb + row) * (HK * DH) + off;
        half_t*       lp = sK + row * SK_STRIDE + off;
#if defined(HAVE_ASYNC_LDS)
        __builtin_amdgcn_global_load_async_to_lds_b128((g1_v4i*)gp, (l3_v4i*)lp, 0, 0);
#else
        *(uint4*)lp = *(const uint4*)gp;
#endif
      }
    }

    // ---- cooperative transpose of V tile (32 x 128) into sVt[d][n]
    {
      const int vrow = tid & 31;          // k position within tile
      const int dc   = tid >> 5;          // 32-dim chunk
      const half_t* vp = vbase + (size_t)(kb + vrow) * (HK * DH) + dc * 32;
      VLoad vl;
      const uint4* vp4 = (const uint4*)vp;
#pragma unroll
      for (int i = 0; i < 4; ++i) vl.u[i] = vp4[i];
#pragma unroll
      for (int i = 0; i < 32; ++i)
        sVt[(dc * 32 + i) * VT_STRIDE + vrow] = vl.e[i];
    }
#if defined(HAVE_ASYNC_LDS)
    __builtin_amdgcn_s_wait_asynccnt(0);
#endif
    __syncthreads();

    // ---- S^T = K_tile * Q^T : two 16x16 tiles over n = [0,16), [16,32), K frags from LDS
    v8f st[2];
    st[0] = (v8f){};
    st[1] = (v8f){};
#pragma unroll
    for (int nt = 0; nt < 2; ++nt) {
      const half_t* krow = sK + (nt * 16 + m) * SK_STRIDE;
#pragma unroll
      for (int dc = 0; dc < 4; ++dc) {
        Frag16 a;
        a.u[0] = *(const uint4*)(krow + dc * 32 + h * 16);
        a.u[1] = *(const uint4*)(krow + dc * 32 + h * 16 + 8);
        st[nt] = __builtin_amdgcn_wmma_f32_16x16x32_f16(
            false, a.h, false, qfrag[dc], (short)0, st[nt], false, false);
      }
    }

    // ---- scale into base-2 domain
#pragma unroll
    for (int nt = 0; nt < 2; ++nt) st[nt] *= SCALE_LOG2E;

    // ---- causal mask: only the <=2 frontier tiles need it (wave-uniform branch)
    if (kt >= nfull) {
#pragma unroll
      for (int nt = 0; nt < 2; ++nt)
#pragma unroll
        for (int r = 0; r < 8; ++r) {
          const int n = kb + nt * 16 + h * 8 + r;
          st[nt][r] = (n <= pos_m) ? st[nt][r] : NEG_INF_F;
        }
    }

    // ---- online softmax (row = m; per-lane n = r + 8h + 16nt)
    float pmax = NEG_INF_F;
#pragma unroll
    for (int nt = 0; nt < 2; ++nt)
#pragma unroll
      for (int r = 0; r < 8; ++r) pmax = fmaxf(pmax, st[nt][r]);
    pmax = fmaxf(pmax, __shfl_xor(pmax, 16));
    const float m_new = fmaxf(row_m, pmax);
    const float corr  = EXP2F(row_m - m_new);

    float psum = 0.0f;
    unsigned pk[2][4];
#pragma unroll
    for (int nt = 0; nt < 2; ++nt)
#pragma unroll
      for (int j = 0; j < 4; ++j) {
        const float e0 = EXP2F(st[nt][2 * j]     - m_new);
        const float e1 = EXP2F(st[nt][2 * j + 1] - m_new);
        psum += e0 + e1;
        pk[nt][j] = pack_h2(e0, e1);
      }
    psum += __shfl_xor(psum, 16);
    row_l = row_l * corr + psum;
    row_m = m_new;

    // ---- build P^T B-fragment: lanes 0-15 hold n=0..15, lanes 16-31 n=16..31
    FragP pf;
#pragma unroll
    for (int j = 0; j < 4; ++j) {
      const unsigned a0 = pk[0][j];             // (n = 2j   + 8h, m)
      const unsigned a1 = pk[1][j];             // (n = 16+2j+ 8h, m)
      const unsigned b0 = __shfl_xor(a0, 16);
      const unsigned b1 = __shfl_xor(a1, 16);
      pf.u[j]     = (h == 0) ? a0 : b1;         // K = 2j,2j+1   | 16+2j,16+2j+1
      pf.u[4 + j] = (h == 0) ? b0 : a1;         // K = 8+2j,...  | 24+2j,...
    }

    // ---- rescale O^T (packed f32 muls), then O^T += V^T * P^T (8 d-tiles)
#pragma unroll
    for (int t = 0; t < 8; ++t) o[t] *= corr;

#pragma unroll
    for (int t = 0; t < 8; ++t) {
      Frag16 a;
      const uint4* r0 = (const uint4*)&sVt[(t * 16 + m) * VT_STRIDE + h * 16];
      const uint4* r1 = (const uint4*)&sVt[(t * 16 + m) * VT_STRIDE + h * 16 + 8];
      a.u[0] = r0[0];
      a.u[1] = r1[0];
      o[t] = __builtin_amdgcn_wmma_f32_16x16x32_f16(
          false, a.h, false, pf.h, (short)0, o[t], false, false);
    }
  }

  // ---- epilogue: out[q_start+m][headq][16t + 8h + r] = o[t][r] / row_l
  const float rl = RCPF(row_l);
  half_t* orow = out + ((size_t)(q_start + m) * HQ + headq) * DH;
#pragma unroll
  for (int t = 0; t < 8; ++t) {
    const v8f ov = o[t] * rl;
    PackO pkd;
#pragma unroll
    for (int r = 0; r < 8; ++r) pkd.e[r] = (half_t)ov[r];
    *(uint4*)(orow + t * 16 + h * 8) = pkd.u;
  }
}

extern "C" void kernel_launch(void* const* d_in, const int* in_sizes, int n_in,
                              void* d_out, int out_size, void* d_ws, size_t ws_size,
                              hipStream_t stream) {
  const half_t* q    = (const half_t*)d_in[0];
  const half_t* kk   = (const half_t*)d_in[1];
  const half_t* vv   = (const half_t*)d_in[2];
  const int*    cu_q = (const int*)d_in[3];
  const int*    cu_k = (const int*)d_in[4];
  const int nseq     = in_sizes[3] - 1;

  const int total_q = in_sizes[0] / (HQ * DH);   // 768
  dim3 grid(total_q / 16, HK);
  dim3 block(128);
  zigzag_flash_attn_kernel<<<grid, block, 0, stream>>>(
      q, kk, vv, cu_q, cu_k, nseq, (half_t*)d_out);
}
